// Ns_Chebnet_71064528880231
// MI455X (gfx1250) — matrix operand order repaired and checked by reading, not compiled, not run on hardware
//
#include <hip/hip_runtime.h>

// ---------------- Problem constants (from reference) ----------------
#define NNODE 100000
#define NEDGE 800000
#define KCHEB 10
#define NLAYER 8
static const int CH[NLAYER + 1] = {2, 32, 64, 128, 256, 128, 64, 32, 1};

#define NROWT 6250                 // ceil(NNODE/16)
#define ROWT_PAD 6256              // rounded up to multiple of 8 waves/block
#define KT_MAX 80                  // KCHEB * 256/32
#define CDIV(a, b) (((a) + (b) - 1) / (b))

typedef __attribute__((ext_vector_type(16))) __bf16 v16bf;
typedef __attribute__((ext_vector_type(8))) float v8f;

// fp32 -> bf16 round-to-nearest-even
static __device__ __forceinline__ unsigned short f2bf(float f) {
  unsigned int u = __float_as_uint(f);
  return (unsigned short)((u + 0x7FFFu + ((u >> 16) & 1u)) >> 16);
}

// ---------------- Elementwise helpers ----------------
__global__ void k_zero(float* __restrict__ p, int n4) {
  int t = blockIdx.x * blockDim.x + threadIdx.x;
  if (t < n4) ((float4*)p)[t] = make_float4(0.f, 0.f, 0.f, 0.f);
}

__global__ void k_negcopy(float* __restrict__ o, const float* __restrict__ i, int n4) {
  int t = blockIdx.x * blockDim.x + threadIdx.x;
  if (t < n4) {
    float4 v = ((const float4*)i)[t];
    ((float4*)o)[t] = make_float4(-v.x, -v.y, -v.z, -v.w);
  }
}

// Gather branch input rows out of interleaved x: out[i][c] = x[rowOff + i*rowStride][c]
__global__ void k_gather_x(const float* __restrict__ x, float* __restrict__ o,
                           int rowStride, int rowOff) {
  int t = blockIdx.x * blockDim.x + threadIdx.x;
  if (t >= NNODE * 2) return;
  int i = t >> 1, c = t & 1;
  o[t] = x[(size_t)(rowOff + i * rowStride) * 2 + c];
}

// ---------------- Graph normalization ----------------
__global__ void k_deg(const int* __restrict__ src, float* __restrict__ deg) {
  int e = blockIdx.x * blockDim.x + threadIdx.x;
  if (e < NEDGE) atomicAdd(deg + src[e], 1.0f);
}

__global__ void k_dis(float* __restrict__ d) {
  int i = blockIdx.x * blockDim.x + threadIdx.x;
  if (i < NNODE) {
    float v = d[i];
    d[i] = (v > 0.f) ? rsqrtf(v) : 0.f;
  }
}

__global__ void k_edgew(const int* __restrict__ src, const int* __restrict__ dst,
                        const float* __restrict__ dis, float* __restrict__ w) {
  int e = blockIdx.x * blockDim.x + threadIdx.x;
  if (e < NEDGE) w[e] = -(dis[src[e]] * dis[dst[e]]);
}

// lhat scatter: out[dst] += scale * w[e] * h[src]  (per channel; ci is power of two)
__global__ void k_scatter(const int* __restrict__ src, const int* __restrict__ dst,
                          const float* __restrict__ w, const float* __restrict__ h,
                          float* __restrict__ out, int ciLog2, float scale) {
  unsigned t = blockIdx.x * blockDim.x + threadIdx.x;
  unsigned total = ((unsigned)NEDGE) << ciLog2;
  if (t >= total) return;
  unsigned e = t >> ciLog2;
  unsigned c = t & ((1u << ciLog2) - 1u);
  int s = src[e], d = dst[e];
  float v = scale * w[e] * h[((size_t)s << ciLog2) + c];
  atomicAdd(out + ((size_t)d << ciLog2) + c, v);
}

// ---------------- Fragment packing (pre-swizzled WMMA layouts) ----------------
// A fragment (16x32 bf16, ISA 7.12.2): lane<16: M=lane, K(hi) = hi<8 ? hi : hi+8
//                                      lane>=16: M=lane-16, K(hi) = hi<8 ? hi+8 : hi+16
// Tile (rowTile, j) stored at ((rowTile*nKT + j)*512 + lane*16) halves -> 32B/lane load.
__global__ void k_pack_a(const float* __restrict__ src, unsigned short* __restrict__ A,
                         int ci, int nkt, int nKT, int k) {
  int tid = blockIdx.x * blockDim.x + threadIdx.x;
  int total = ROWT_PAD * nkt * 32;
  if (tid >= total) return;
  int lane = tid & 31;
  int tile = tid >> 5;
  int rowTile = tile / nkt;
  int jj = tile - rowTile * nkt;
  int j = k * nkt + jj;
  int M = rowTile * 16 + (lane & 15);
  unsigned short vals[16];
#pragma unroll
  for (int hi = 0; hi < 16; ++hi) {
    int K = (lane < 16) ? ((hi < 8) ? hi : hi + 8) : ((hi < 8) ? hi + 8 : hi + 16);
    int c = jj * 32 + K;
    float f = (M < NNODE && c < ci) ? src[(size_t)M * ci + c] : 0.f;
    vals[hi] = f2bf(f);
  }
  unsigned short* d = A + ((size_t)rowTile * nKT + j) * 512 + lane * 16;
  ((uint4*)d)[0] = ((const uint4*)vals)[0];
  ((uint4*)d)[1] = ((const uint4*)vals)[1];
}

// B fragment (32x16 bf16): lane<16: N=lane, K(hi) = hi<8 ? hi : hi+8 ; lane>=16: N=lane-16, K(hi) = hi<8 ? hi+8 : hi+16
// Fused weight: Wcat[kk = k*ci_pad + c][o], zero padded. Tile (nt, j) at (nt*nKT + j)*512.
__global__ void k_pack_w(const float* __restrict__ W, unsigned short* __restrict__ Bm,
                         int ci, int co, int ciPadLog2, int nKT, int nNT) {
  int tid = blockIdx.x * blockDim.x + threadIdx.x;
  int total = nNT * nKT * 32;
  if (tid >= total) return;
  int lane = tid & 31;
  int tile = tid >> 5;
  int nt = tile / nKT;
  int j = tile - nt * nKT;
  int N = nt * 16 + (lane & 15);
  int mask = (1 << ciPadLog2) - 1;
  unsigned short vals[16];
#pragma unroll
  for (int hi = 0; hi < 16; ++hi) {
    int K = (lane < 16) ? ((hi < 8) ? hi : hi + 8) : ((hi < 8) ? hi + 8 : hi + 16);
    int kk = j * 32 + K;
    int kidx = kk >> ciPadLog2;
    int c = kk & mask;
    float f = (c < ci && N < co) ? W[((size_t)kidx * ci + c) * co + N] : 0.f;
    vals[hi] = f2bf(f);
  }
  unsigned short* d = Bm + ((size_t)nt * nKT + j) * 512 + lane * 16;
  ((uint4*)d)[0] = ((const uint4*)vals)[0];
  ((uint4*)d)[1] = ((const uint4*)vals)[1];
}

// ---------------- WMMA GEMM: out[n,co] = A[n,32*nKT] @ B[32*nKT,co_pad] + bias ----------
// One wave = one 16-row tile, NT column tiles (fp32 accumulators), K streamed once.
template <int NT>
__global__ void k_gemm(const unsigned short* __restrict__ A,
                       const unsigned short* __restrict__ Bm,
                       const float* __restrict__ bias, float* __restrict__ out,
                       int co, int nKT, int outStride, int relu) {
  int lane = threadIdx.x & 31;
  int rowTile = blockIdx.x * 8 + (threadIdx.x >> 5);  // grid sized so rowTile < ROWT_PAD
  v8f acc[NT] = {};
  const unsigned short* aP = A + ((size_t)rowTile * nKT) * 512 + lane * 16;
  for (int j = 0; j < nKT; ++j) {
    v16bf a = *(const v16bf*)aP;
    __builtin_prefetch(aP + 512, 0, 1);  // global_prefetch_b8 on A stream
    const unsigned short* bP = Bm + (size_t)j * 512 + lane * 16;
#pragma unroll
    for (int t = 0; t < NT; ++t) {
      v16bf b = *(const v16bf*)(bP + (size_t)t * nKT * 512);
      acc[t] = __builtin_amdgcn_wmma_f32_16x16x32_bf16(false, a, false, b, (short)0,
                                                       acc[t], false, false);
    }
    aP += 512;
  }
  // C/D layout: VGPR r: lanes 0-15 -> (M=r, N=lane), lanes 16-31 -> (M=8+r, N=lane-16)
  int rowBase = rowTile * 16 + ((lane >= 16) ? 8 : 0);
  int colLane = lane & 15;
#pragma unroll
  for (int t = 0; t < NT; ++t) {
    int col = t * 16 + colLane;
    if (col < co) {
      float bv = bias[col];
#pragma unroll
      for (int r = 0; r < 8; ++r) {
        int row = rowBase + r;
        if (row < NNODE) {
          float v = acc[t][r] + bv;
          if (relu) v = fmaxf(v, 0.0f);
          out[(size_t)row * outStride + col] = v;
        }
      }
    }
  }
}

static void launch_gemm(int NT, const unsigned short* A, const unsigned short* B,
                        const float* bias, float* out, int co, int nKT, int outStride,
                        int relu, hipStream_t s) {
  dim3 g(ROWT_PAD / 8), b(256);
  switch (NT) {
    case 1:  k_gemm<1><<<g, b, 0, s>>>(A, B, bias, out, co, nKT, outStride, relu); break;
    case 2:  k_gemm<2><<<g, b, 0, s>>>(A, B, bias, out, co, nKT, outStride, relu); break;
    case 4:  k_gemm<4><<<g, b, 0, s>>>(A, B, bias, out, co, nKT, outStride, relu); break;
    case 8:  k_gemm<8><<<g, b, 0, s>>>(A, B, bias, out, co, nKT, outStride, relu); break;
    default: k_gemm<16><<<g, b, 0, s>>>(A, B, bias, out, co, nKT, outStride, relu); break;
  }
}

// ---------------- One ChebNet branch ----------------
static void run_branch(const float* x, int rowStride, int rowOff, const int* src,
                       const int* dst, void* const* d_in, int pbase, float* finalOut,
                       int finalStride, float* F0, float* F1, float* F2, float* F3,
                       float* dis, float* wbuf, unsigned short* Abig,
                       unsigned short* Wpad, hipStream_t st) {
  const int B = 256;
  // deg -> dis -> edge weights
  k_zero<<<CDIV(NNODE / 4, B), B, 0, st>>>(dis, NNODE / 4);
  k_deg<<<CDIV(NEDGE, B), B, 0, st>>>(src, dis);
  k_dis<<<CDIV(NNODE, B), B, 0, st>>>(dis);
  k_edgew<<<CDIV(NEDGE, B), B, 0, st>>>(src, dst, dis, wbuf);

  float* F[4] = {F0, F1, F2, F3};
  int cur = 0;
  k_gather_x<<<CDIV(NNODE * 2, B), B, 0, st>>>(x, F[0], rowStride, rowOff);

  for (int L = 0; L < NLAYER; ++L) {
    int ci = CH[L], co = CH[L + 1];
    int ciPad = ci < 32 ? 32 : ci;
    int ciLog2 = __builtin_ctz(ci);
    int ciPadLog2 = __builtin_ctz(ciPad);
    int nkt = ciPad >> 5;           // K-tiles per Chebyshev term
    int nKT = KCHEB * nkt;          // total K-tiles in fused GEMM
    int NT = (co < 16 ? 16 : co) >> 4;
    const float* W = (const float*)d_in[pbase + 2 * L];
    const float* bias = (const float*)d_in[pbase + 2 * L + 1];

    k_pack_w<<<CDIV(NT * nKT * 32, B), B, 0, st>>>(W, Wpad, ci, co, ciPadLog2, nKT, NT);

    float* T0 = F[cur];
    float* T1 = F[(cur + 1) & 3];
    float* T2 = F[(cur + 2) & 3];
    float* Ob = F[(cur + 3) & 3];
    int nc4 = (NNODE * ci) / 4;
    int packTotal = ROWT_PAD * nkt * 32;
    unsigned tot = ((unsigned)NEDGE) << ciLog2;

    // k = 0: Tx0 = layer input
    k_pack_a<<<CDIV(packTotal, B), B, 0, st>>>(T0, Abig, ci, nkt, nKT, 0);
    // k = 1: Tx1 = lhat(Tx0)
    k_zero<<<CDIV(nc4, B), B, 0, st>>>(T1, nc4);
    k_scatter<<<CDIV(tot, B), B, 0, st>>>(src, dst, wbuf, T0, T1, ciLog2, 1.0f);
    k_pack_a<<<CDIV(packTotal, B), B, 0, st>>>(T1, Abig, ci, nkt, nKT, 1);
    // k >= 2: Tx2 = 2*lhat(Tx1) - Tx0  (init target to -Tx0, scatter with scale 2)
    for (int k = 2; k < KCHEB; ++k) {
      k_negcopy<<<CDIV(nc4, B), B, 0, st>>>(T2, T0, nc4);
      k_scatter<<<CDIV(tot, B), B, 0, st>>>(src, dst, wbuf, T1, T2, ciLog2, 2.0f);
      k_pack_a<<<CDIV(packTotal, B), B, 0, st>>>(T2, Abig, ci, nkt, nKT, k);
      float* tmp = T0; T0 = T1; T1 = T2; T2 = tmp;
    }

    int relu = (L < NLAYER - 1) ? 1 : 0;
    float* outPtr = (L == NLAYER - 1) ? finalOut : Ob;
    int outStride = (L == NLAYER - 1) ? finalStride : co;
    launch_gemm(NT, Abig, Wpad, bias, outPtr, co, nKT, outStride, relu, st);
    cur = (cur + 3) & 3;  // next layer's input is Ob
  }
}

// ---------------- Entry point ----------------
extern "C" void kernel_launch(void* const* d_in, const int* in_sizes, int n_in,
                              void* d_out, int out_size, void* d_ws, size_t ws_size,
                              hipStream_t stream) {
  (void)in_sizes; (void)n_in; (void)out_size; (void)ws_size;
  const float* x = (const float*)d_in[0];
  const int* ei = (const int*)d_in[1];  // (2, 3*NEDGE) row-major

  // Carve workspace (256B aligned)
  char* p = (char*)d_ws;
  auto carve = [&](size_t bytes) -> void* {
    void* r = (void*)p;
    p += (bytes + 255) & ~(size_t)255;
    return r;
  };
  unsigned short* Abig = (unsigned short*)carve((size_t)ROWT_PAD * KT_MAX * 1024);
  unsigned short* Wpad = (unsigned short*)carve((size_t)16 * KT_MAX * 1024);
  float* F0 = (float*)carve((size_t)NNODE * 256 * 4);
  float* F1 = (float*)carve((size_t)NNODE * 256 * 4);
  float* F2 = (float*)carve((size_t)NNODE * 256 * 4);
  float* F3 = (float*)carve((size_t)NNODE * 256 * 4);
  float* dis = (float*)carve((size_t)NNODE * 4);
  float* wbuf = (float*)carve((size_t)NEDGE * 4);

  const int* src1 = ei + 0 * NEDGE;
  const int* src2 = ei + 1 * NEDGE;
  const int* src3 = ei + 2 * NEDGE;
  const int* dst1 = ei + 3 * NEDGE + 0 * NEDGE;
  const int* dst2 = ei + 3 * NEDGE + 1 * NEDGE;
  const int* dst3 = ei + 3 * NEDGE + 2 * NEDGE;

  float* out = (float*)d_out;
  // y1 -> out[2i], y2 -> out[2i+1], y3 -> out[2*N1 + i]
  run_branch(x, 2, 0,         src1, dst1, d_in, 5,  out,             2, F0, F1, F2, F3, dis, wbuf, Abig, Wpad, stream);
  run_branch(x, 2, 1,         src2, dst2, d_in, 21, out + 1,         2, F0, F1, F2, F3, dis, wbuf, Abig, Wpad, stream);
  run_branch(x, 1, 2 * NNODE, src3, dst3, d_in, 37, out + 2 * NNODE, 1, F0, F1, F2, F3, dis, wbuf, Abig, Wpad, stream);
}
